// Transformer_74062416053365
// MI455X (gfx1250) — compile-verified, matrix operand order checked by reading, not run on hardware
//
#include <hip/hip_runtime.h>
#include <hip/hip_bf16.h>
#include <cstdint>
#include <cstddef>

// ---------------------------------------------------------------------------
// CDNA5 / gfx1250 WMMA primitives (wave32)
// ---------------------------------------------------------------------------
typedef __attribute__((ext_vector_type(16))) _Float16 v16h;
typedef __attribute__((ext_vector_type(8)))  _Float16 v8h;
typedef __attribute__((ext_vector_type(4)))  _Float16 v4h;
typedef __attribute__((ext_vector_type(8)))  float    v8f;

#define WMMA_F32_F16(a, b, c) \
  __builtin_amdgcn_wmma_f32_16x16x32_f16(false, (a), false, (b), (short)0, (c), false, false)

__device__ __forceinline__ v8f zero8() {
  v8f z;
#pragma unroll
  for (int i = 0; i < 8; ++i) z[i] = 0.0f;
  return z;
}

// Fragment loader for row-major LDS storage [r][k] (stride in halves, all
// strides/offsets multiples of 8 halves => 16B-aligned b128 LDS loads).
// Lane holds r = lane&15; k = (ii>>3)*16 + (lane>>4)*8 + (ii&7)  (ISA 7.12.2).
// Used for A operands ([m][k]) and for B operands stored transposed ([n][k]).
__device__ __forceinline__ v16h frag_rm(const _Float16* base, int stride, int lane) {
  const _Float16* p = base + (lane & 15) * stride + ((lane >> 4) << 3);
  v8h lo = *(const v8h*)(p);
  v8h hi = *(const v8h*)(p + 16);
  return __builtin_shufflevector(lo, hi, 0, 1, 2, 3, 4, 5, 6, 7,
                                 8, 9, 10, 11, 12, 13, 14, 15);
}

// D-tile (16x16 f32) store to transposed f16 storage [col][row]:
// per lane col = colBase+(lane&15) fixed, rows contiguous -> one b128 store.
__device__ __forceinline__ void store_tile_T(v8f c, _Float16* dst, int stride,
                                             int colBase, int rowBase, int lane) {
  v8h h;
#pragma unroll
  for (int j = 0; j < 8; ++j) h[j] = (_Float16)c[j];
  *(v8h*)(dst + (colBase + (lane & 15)) * stride + rowBase + ((lane >> 4) << 3)) = h;
}

// D-tile store to natural f16 storage [row][col] (scalar stores).
__device__ __forceinline__ void store_tile_N(v8f c, _Float16* dst, int stride,
                                             int rowBase, int colBase, int lane) {
  int col = colBase + (lane & 15), rb = rowBase + ((lane >> 4) << 3);
#pragma unroll
  for (int j = 0; j < 8; ++j) dst[(rb + j) * stride + col] = (_Float16)c[j];
}

// ---------------------------------------------------------------------------
// Generic WMMA GEMM:  C[M,N] = act(A[M,K] @ B[K,N] + bias) + res
// grid = (N/16, M/64), block = 128 threads (4 waves, one 16x16 tile each)
// act: 0=none, 1=relu, 2=tanh.  M multiple of 64, N multiple of 16.
// ---------------------------------------------------------------------------
__global__ __launch_bounds__(128) void k_gemm(
    const float* __restrict__ A, int lda,
    const float* __restrict__ B, int ldb,
    const float* __restrict__ bias,
    const float* __restrict__ res,
    float* __restrict__ C, int ldc,
    int M, int K, int N, int act) {
  const int AS = 40, BS = 40;
  __shared__ _Float16 Ap[64 * 40];      // A panel  [row][k]
  __shared__ _Float16 BpT[16 * 40];     // B panel transposed [n][k]
  int tid = threadIdx.x, lane = tid & 31, wid = tid >> 5;
  int r0 = blockIdx.y * 64, c0 = blockIdx.x * 16;
  v8f acc = zero8();
  for (int ks = 0; ks < K; ks += 32) {
    if (ks + 32 < K)  // gfx1250 global_prefetch_b8 for next A panel
      __builtin_prefetch(A + (size_t)(r0 + (tid >> 1)) * lda + ks + 32, 0, 1);
    if (ks + 32 <= K) {
      // fast path: batched float4 loads (all K>=32 call sites are 16B aligned)
      int r = tid >> 1, cc = (tid & 1) * 16;
      const float4* p = (const float4*)(A + (size_t)(r0 + r) * lda + ks + cc);
      float4 f0 = p[0], f1 = p[1], f2 = p[2], f3 = p[3];
      v8h h0, h1;
      h0[0] = (_Float16)f0.x; h0[1] = (_Float16)f0.y; h0[2] = (_Float16)f0.z; h0[3] = (_Float16)f0.w;
      h0[4] = (_Float16)f1.x; h0[5] = (_Float16)f1.y; h0[6] = (_Float16)f1.z; h0[7] = (_Float16)f1.w;
      h1[0] = (_Float16)f2.x; h1[1] = (_Float16)f2.y; h1[2] = (_Float16)f2.z; h1[3] = (_Float16)f2.w;
      h1[4] = (_Float16)f3.x; h1[5] = (_Float16)f3.y; h1[6] = (_Float16)f3.z; h1[7] = (_Float16)f3.w;
      *(v8h*)(Ap + r * AS + cc) = h0;
      *(v8h*)(Ap + r * AS + cc + 8) = h1;
    } else {
      // tail path (K=12 etc.): elementwise guarded
      for (int i = tid; i < 64 * 32; i += 128) {
        int r = i >> 5, c = i & 31;
        int gk = ks + c;
        float v = (gk < K) ? A[(size_t)(r0 + r) * lda + gk] : 0.0f;
        Ap[r * AS + c] = (_Float16)v;
      }
    }
    {  // B panel, transposed gather: [n][k]
      int n = tid >> 3, kc = (tid & 7) * 4;
      v4h h;
#pragma unroll
      for (int i = 0; i < 4; ++i) {
        int gk = ks + kc + i;
        h[i] = (_Float16)((gk < K) ? B[(size_t)gk * ldb + c0 + n] : 0.0f);
      }
      *(v4h*)(BpT + n * BS + kc) = h;
    }
    __syncthreads();
    v16h af = frag_rm(Ap + (wid * 16) * AS, AS, lane);
    v16h bf = frag_rm(BpT, BS, lane);
    acc = WMMA_F32_F16(af, bf, acc);
    __syncthreads();
  }
  int hl = lane >> 4, nl = lane & 15;
  int col = c0 + nl;
#pragma unroll
  for (int j = 0; j < 8; ++j) {
    int row = r0 + wid * 16 + j + 8 * hl;
    float v = acc[j];
    if (bias) v += bias[col];
    if (act == 1) v = fmaxf(v, 0.0f);
    else if (act == 2) v = tanhf(v);
    if (res) v += res[(size_t)row * ldc + col];
    C[(size_t)row * ldc + col] = v;
  }
}

// ---------------------------------------------------------------------------
// Fused block attention (one coord-sort pass): for each 128-row sorted block,
// per head: K, V(T), Q GEMMs -> scores WMMA + RPE bias -> register softmax
// (wave32 shuffle reductions) -> O WMMA -> O @ wo accumulated in WMMA C.
// Scatters h[g] += 0.5 * (o @ wo).  block = 256 threads (8 waves x 16 rows).
// ---------------------------------------------------------------------------
#define XH_S 72    // xh  [128 row][64 k]      natural (A op)
#define W_S  72    // wslT[64 n][64 k]         transposed weights (B op)
#define KH_S 72    // kh  [128 key][64 dim]    natural == [n][k] for Q.K^T (B op)
#define VT_S 136   // vhT [64 dim][128 key]    transposed V (B op for attn.V)
#define Q_S  72    // qh  [128 row][64 dim]    natural (A op)
#define AT_S 136   // at  [128 row][128 key]   natural (A op)
#define O_S  72    // osc [128 row][64 dim]    natural (A op)
#define ATT_SMEM_BYTES \
  ((128 * XH_S + 64 * W_S + 128 * KH_S + 64 * VT_S + 128 * Q_S + 128 * AT_S + 128 * O_S) * 2 + \
   128 * 2 * 4)

__global__ __launch_bounds__(256) void k_attention(
    const float* __restrict__ xn, const float* __restrict__ coords2,
    const int* __restrict__ srt,
    const float* __restrict__ wq, const float* __restrict__ wk,
    const float* __restrict__ wvp, const float* __restrict__ wo,
    const float* __restrict__ w_rpe, const float* __restrict__ w_rpe_b,
    float* __restrict__ h) {
  extern __shared__ char smem[];
  _Float16* xh   = (_Float16*)smem;
  _Float16* wslT = xh + 128 * XH_S;
  _Float16* kh   = wslT + 64 * W_S;
  _Float16* vhT  = kh + 128 * KH_S;
  _Float16* qh   = vhT + 64 * VT_S;
  _Float16* at   = qh + 128 * Q_S;
  _Float16* osc  = at + 128 * AT_S;
  float*    cl   = (float*)(osc + 128 * O_S);  // 128 x 2 coords

  int tid = threadIdx.x, lane = tid & 31, wid = tid >> 5;
  int blk = blockIdx.x;
  int hl = lane >> 4, nl = lane & 15;

  // gather LN'd rows through the sort permutation: 16-half chunks, b128 stores
#pragma unroll
  for (int q = 0; q < 2; ++q) {
    int chunk = tid + q * 256;                 // 512 chunks of 16 halves
    int r = chunk >> 2, cc = (chunk & 3) * 16;
    int g = srt[blk * 128 + r];
    const float4* p = (const float4*)(xn + (size_t)g * 64 + cc);
    float4 f0 = p[0], f1 = p[1], f2 = p[2], f3 = p[3];
    v8h h0, h1;
    h0[0] = (_Float16)f0.x; h0[1] = (_Float16)f0.y; h0[2] = (_Float16)f0.z; h0[3] = (_Float16)f0.w;
    h0[4] = (_Float16)f1.x; h0[5] = (_Float16)f1.y; h0[6] = (_Float16)f1.z; h0[7] = (_Float16)f1.w;
    h1[0] = (_Float16)f2.x; h1[1] = (_Float16)f2.y; h1[2] = (_Float16)f2.z; h1[3] = (_Float16)f2.w;
    h1[4] = (_Float16)f3.x; h1[5] = (_Float16)f3.y; h1[6] = (_Float16)f3.z; h1[7] = (_Float16)f3.w;
    *(v8h*)(xh + r * XH_S + cc) = h0;
    *(v8h*)(xh + r * XH_S + cc + 8) = h1;
  }
  for (int i = tid; i < 128 * 2; i += 256) {
    int r = i >> 1, c = i & 1;
    int g = srt[blk * 128 + r];
    cl[i] = coords2[(size_t)g * 2 + c];
  }
  __syncthreads();

  // stage a 64x64 weight slice TRANSPOSED into wslT[n][k]
  auto stageWT = [&](const float* W, int rowBase, int rowStride, int colBase) {
    int c = tid & 63, rchunk = (tid >> 6) * 16;  // 64 cols x 4 row-chunks
    float t[16];
#pragma unroll
    for (int i = 0; i < 16; ++i)
      t[i] = W[(size_t)(rowBase + rchunk + i) * rowStride + colBase + c];
    v8h h0, h1;
#pragma unroll
    for (int i = 0; i < 8; ++i) { h0[i] = (_Float16)t[i]; h1[i] = (_Float16)t[8 + i]; }
    *(v8h*)(wslT + c * W_S + rchunk) = h0;
    *(v8h*)(wslT + c * W_S + rchunk + 8) = h1;
  };

  // xh[128x64] @ wslT -> dst; wave wid owns row-tile mt==wid, 4 col tiles
  auto gemmXW = [&](_Float16* dst, int dstStride, bool transposed) {
#pragma unroll
    for (int nt = 0; nt < 4; ++nt) {
      v8f c = zero8();
      c = WMMA_F32_F16(frag_rm(xh + wid * 16 * XH_S, XH_S, lane),
                       frag_rm(wslT + nt * 16 * W_S, W_S, lane), c);
      c = WMMA_F32_F16(frag_rm(xh + wid * 16 * XH_S + 32, XH_S, lane),
                       frag_rm(wslT + nt * 16 * W_S + 32, W_S, lane), c);
      if (transposed) store_tile_T(c, dst, dstStride, nt * 16, wid * 16, lane);
      else            store_tile_N(c, dst, dstStride, wid * 16, nt * 16, lane);
    }
  };

  v8f accD[4];
#pragma unroll
  for (int ot = 0; ot < 4; ++ot) accD[ot] = zero8();

  for (int hd = 0; hd < 8; ++hd) {
    // RPE weights: feats order [de, dp, de^2, dp^2, de^3, dp^3]
    float rw0 = w_rpe[hd],      rw1 = w_rpe[8 + hd],  rw2 = w_rpe[16 + hd];
    float rw3 = w_rpe[24 + hd], rw4 = w_rpe[32 + hd], rw5 = w_rpe[40 + hd];
    float rbb = w_rpe_b[hd];

    stageWT(wk, 0, 512, hd * 64);  __syncthreads();
    gemmXW(kh, KH_S, false);       __syncthreads();   // K natural = [n][k] for scores
    stageWT(wvp, 0, 512, hd * 64); __syncthreads();
    gemmXW(vhT, VT_S, true);       __syncthreads();   // V transposed = [n][k] for O
    stageWT(wq, 0, 512, hd * 64);  __syncthreads();
    gemmXW(qh, Q_S, false);        __syncthreads();

    // ---- scores: Q(16x64) @ K^T(64x128) in registers, + RPE bias ----
    v16h qa0 = frag_rm(qh + wid * 16 * Q_S, Q_S, lane);
    v16h qa1 = frag_rm(qh + wid * 16 * Q_S + 32, Q_S, lane);
    v8f sc[8];
#pragma unroll
    for (int ct = 0; ct < 8; ++ct) {
      v8f c = zero8();
      c = WMMA_F32_F16(qa0, frag_rm(kh + ct * 16 * KH_S, KH_S, lane), c);
      c = WMMA_F32_F16(qa1, frag_rm(kh + ct * 16 * KH_S + 32, KH_S, lane), c);
      float cjx = cl[(ct * 16 + nl) * 2 + 0], cjy = cl[(ct * 16 + nl) * 2 + 1];
#pragma unroll
      for (int j = 0; j < 8; ++j) {
        int ir = wid * 16 + j + 8 * hl;
        float de = fabsf(cl[ir * 2 + 0] - cjx);
        float dp = fabsf(cl[ir * 2 + 1] - cjy);
        float de2 = de * de, dp2 = dp * dp;
        float bias = rbb + rw0 * de + rw1 * dp + rw2 * de2 + rw3 * dp2
                         + rw4 * de2 * de + rw5 * dp2 * dp;
        c[j] = c[j] * 0.125f + bias;  // 1/sqrt(64)
      }
      sc[ct] = c;
    }
    // ---- register softmax over 128 keys (cols in 16 lanes x 8 tiles) ----
    float mrow[8], srow[8];
#pragma unroll
    for (int j = 0; j < 8; ++j) {
      float m = -3.0e38f;
#pragma unroll
      for (int ct = 0; ct < 8; ++ct) m = fmaxf(m, sc[ct][j]);
#pragma unroll
      for (int o = 1; o < 16; o <<= 1) m = fmaxf(m, __shfl_xor(m, o, 32));
      mrow[j] = m;
    }
#pragma unroll
    for (int j = 0; j < 8; ++j) {
      float s = 0.0f;
#pragma unroll
      for (int ct = 0; ct < 8; ++ct) {
        float e = __expf(sc[ct][j] - mrow[j]);
        sc[ct][j] = e;
        s += e;
      }
#pragma unroll
      for (int o = 1; o < 16; o <<= 1) s += __shfl_xor(s, o, 32);
      srow[j] = 1.0f / s;
    }
#pragma unroll
    for (int ct = 0; ct < 8; ++ct)
#pragma unroll
      for (int j = 0; j < 8; ++j)
        at[(wid * 16 + j + 8 * hl) * AT_S + ct * 16 + nl] =
            (_Float16)(sc[ct][j] * srow[j]);

    // ---- O = attn(16x128) @ V(128x64) ----
    v16h aat[4];
#pragma unroll
    for (int kk = 0; kk < 4; ++kk)
      aat[kk] = frag_rm(at + wid * 16 * AT_S + kk * 32, AT_S, lane);
#pragma unroll
    for (int ot = 0; ot < 4; ++ot) {
      v8f o = zero8();
#pragma unroll
      for (int kk = 0; kk < 4; ++kk)
        o = WMMA_F32_F16(aat[kk], frag_rm(vhT + ot * 16 * VT_S + kk * 32, VT_S, lane), o);
      store_tile_N(o, osc, O_S, wid * 16, ot * 16, lane);
    }
    __syncthreads();
    // ---- accumulate O @ wo_slice into persistent WMMA C accumulators ----
    stageWT(wo, hd * 64, 64, 0);
    __syncthreads();
    v16h oa0 = frag_rm(osc + wid * 16 * O_S, O_S, lane);
    v16h oa1 = frag_rm(osc + wid * 16 * O_S + 32, O_S, lane);
#pragma unroll
    for (int ot = 0; ot < 4; ++ot) {
      accD[ot] = WMMA_F32_F16(oa0, frag_rm(wslT + ot * 16 * W_S, W_S, lane), accD[ot]);
      accD[ot] = WMMA_F32_F16(oa1, frag_rm(wslT + ot * 16 * W_S + 32, W_S, lane), accD[ot]);
    }
    __syncthreads();
  }

  // scatter: h[g] += 0.5 * (o @ wo)   (each row unique within this launch)
#pragma unroll
  for (int ot = 0; ot < 4; ++ot)
#pragma unroll
    for (int j = 0; j < 8; ++j) {
      int row = wid * 16 + j + 8 * hl;
      int g = srt[blk * 128 + row];
      int col = ot * 16 + nl;
      float* p = h + (size_t)g * 64 + col;
      *p += 0.5f * accD[ot][j];
    }
}

// ---------------------------------------------------------------------------
// LayerNorm (wave32 per row, shuffle reductions).  act: 0=none, 1=tanh out.
// ---------------------------------------------------------------------------
__global__ __launch_bounds__(256) void k_layernorm(
    const float* __restrict__ x, const float* __restrict__ g,
    const float* __restrict__ b, float* __restrict__ y,
    int rows, int dim, int act) {
  int wid = threadIdx.x >> 5, lane = threadIdx.x & 31;
  int row = blockIdx.x * 8 + wid;
  if (row >= rows) return;
  const float* xr = x + (size_t)row * dim;
  float s = 0.0f, s2 = 0.0f;
  for (int c = lane; c < dim; c += 32) { float v = xr[c]; s += v; s2 += v * v; }
  for (int o = 16; o > 0; o >>= 1) { s += __shfl_xor(s, o, 32); s2 += __shfl_xor(s2, o, 32); }
  float m = s / dim;
  float r = rsqrtf(s2 / dim - m * m + 1e-5f);
  float* yr = y + (size_t)row * dim;
  for (int c = lane; c < dim; c += 32) {
    float v = (xr[c] - m) * r * g[c] + b[c];
    if (act) v = tanhf(v);
    yr[c] = v;
  }
}

// ---------------------------------------------------------------------------
// Small helper kernels
// ---------------------------------------------------------------------------
__global__ void k_pad(const float* __restrict__ x, const float* __restrict__ coords,
                      float* __restrict__ xp, float* __restrict__ c2, int N, int Np) {
  int i = blockIdx.x * blockDim.x + threadIdx.x;
  if (i < Np * 12) { int r = i / 12; xp[i] = (r < N) ? x[i] : 0.0f; }
  if (i < Np * 2) { int r = i >> 1, c = i & 1; c2[i] = (r < N) ? coords[r * 3 + c] : 0.0f; }
}

__global__ void k_sortinit(float* __restrict__ key, int* __restrict__ idx,
                           const float* __restrict__ coords, int N, int Np, int NS, int d) {
  int i = blockIdx.x * blockDim.x + threadIdx.x;
  if (i >= NS) return;
  idx[i] = i;
  key[i] = (i < N) ? coords[(size_t)i * 3 + d] : ((i < Np) ? 1.0e9f : 3.0e38f);
}

__global__ void k_bitonic(float* __restrict__ key, int* __restrict__ idx,
                          int j, int k, int n) {
  int i = blockIdx.x * blockDim.x + threadIdx.x;
  if (i >= n) return;
  int ixj = i ^ j;
  if (ixj <= i) return;
  bool up = ((i & k) == 0);
  float a = key[i], b = key[ixj];
  if (up ? (a > b) : (a < b)) {
    key[i] = b; key[ixj] = a;
    int t = idx[i]; idx[i] = idx[ixj]; idx[ixj] = t;
  }
}

__global__ void k_take(int* __restrict__ srt, const int* __restrict__ sidx, int Np) {
  int i = blockIdx.x * blockDim.x + threadIdx.x;
  if (i < Np) srt[i] = sidx[i];
}

__global__ void k_copy_col(const float* __restrict__ h, float* __restrict__ allh,
                           int Np, int off) {
  int i = blockIdx.x * blockDim.x + threadIdx.x;
  if (i < Np * 64) {
    int r = i >> 6, c = i & 63;
    allh[(size_t)r * 320 + off + c] = h[i];
  }
}

__global__ void k_zero(float* __restrict__ p, int n) {
  int i = blockIdx.x * blockDim.x + threadIdx.x;
  if (i < n) p[i] = 0.0f;
}

__global__ void k_pool(const float* __restrict__ enc, const float* __restrict__ z,
                       const int* __restrict__ batch, float* __restrict__ psum,
                       float* __restrict__ pcnt, int N) {
  int i = blockIdx.x * blockDim.x + threadIdx.x;
  if (i >= N * 32) return;
  int r = i >> 5, c = i & 31;
  int g = batch[r];
  atomicAdd(&psum[g * 32 + c], enc[i] + z[i]);
  if (c == 0) atomicAdd(&pcnt[g], 1.0f);
}

__global__ void k_final(const float* __restrict__ psum, const float* __restrict__ pcnt,
                        const float* __restrict__ opw, const float* __restrict__ opb,
                        float* __restrict__ out, int G) {
  int g = blockIdx.x * blockDim.x + threadIdx.x;
  if (g >= G) return;
  float cnt = fmaxf(pcnt[g], 1.0f);
  float acc = opb[0];
  for (int c = 0; c < 32; ++c) acc += (psum[g * 32 + c] / cnt) * opw[c];
  out[g] = acc;
}

// ---------------------------------------------------------------------------
// Host orchestration
// ---------------------------------------------------------------------------
extern "C" void kernel_launch(void* const* d_in, const int* in_sizes, int n_in,
                              void* d_out, int out_size, void* d_ws, size_t ws_size,
                              hipStream_t stream) {
  // ---- input unpack (jax pytree / sorted-key flattening of the params dict) ----
  const float* x      = (const float*)d_in[0];
  const float* coords = (const float*)d_in[1];
  const int*   batch  = (const int*)d_in[2];
  const float* Ww     = (const float*)d_in[3];   // [320,32]
  const float* fe1b   = (const float*)d_in[4];
  const float* fe1w   = (const float*)d_in[5];   // [12,64]
  const float* fe2b   = (const float*)d_in[6];
  const float* fe2w   = (const float*)d_in[7];   // [64,64]
  struct LayerP {
    const float *f1b, *f1w, *f2b, *f2w, *n1b, *n1g, *n2b, *n2g,
                *wrpe, *wrpeb, *wk, *wo, *wq, *wv;
  };
  LayerP L[4];
  for (int l = 0; l < 4; ++l) {
    int b0 = 8 + l * 14;  // sorted: f1b,f1w,f2b,f2w,n1b,n1g,n2b,n2g,w_rpe,w_rpe_b,wk,wo,wq,wv
    L[l].f1b   = (const float*)d_in[b0 + 0];
    L[l].f1w   = (const float*)d_in[b0 + 1];
    L[l].f2b   = (const float*)d_in[b0 + 2];
    L[l].f2w   = (const float*)d_in[b0 + 3];
    L[l].n1b   = (const float*)d_in[b0 + 4];
    L[l].n1g   = (const float*)d_in[b0 + 5];
    L[l].n2b   = (const float*)d_in[b0 + 6];
    L[l].n2g   = (const float*)d_in[b0 + 7];
    L[l].wrpe  = (const float*)d_in[b0 + 8];
    L[l].wrpeb = (const float*)d_in[b0 + 9];
    L[l].wk    = (const float*)d_in[b0 + 10];
    L[l].wo    = (const float*)d_in[b0 + 11];
    L[l].wq    = (const float*)d_in[b0 + 12];
    L[l].wv    = (const float*)d_in[b0 + 13];
  }
  const float *mw[5], *mb[5], *mg[4], *mnb[4];
  for (int i = 0; i < 4; ++i) {  // sorted: b,g,nb,w
    int b0 = 64 + i * 4;
    mb[i]  = (const float*)d_in[b0 + 0];
    mg[i]  = (const float*)d_in[b0 + 1];
    mnb[i] = (const float*)d_in[b0 + 2];
    mw[i]  = (const float*)d_in[b0 + 3];
  }
  mb[4] = (const float*)d_in[80];
  mw[4] = (const float*)d_in[81];
  const float* opb = (const float*)d_in[82];
  const float* opw = (const float*)d_in[83];
  (void)n_in; (void)out_size; (void)ws_size;

  const int N  = in_sizes[0] / 12;
  const int Np = ((N + 127) / 128) * 128;
  int NS = 1; while (NS < Np) NS <<= 1;
  const int nb = Np / 128;
  const int G = 200;

  // ---- workspace partition ----
  char* base = (char*)d_ws;
  auto alloc = [&](size_t bytes) {
    void* p = (void*)base;
    base += (bytes + 255) & ~(size_t)255;
    return p;
  };
  float* xp   = (float*)alloc((size_t)Np * 12 * 4);
  float* c2   = (float*)alloc((size_t)Np * 2 * 4);
  float* h    = (float*)alloc((size_t)Np * 64 * 4);
  float* xn   = (float*)alloc((size_t)Np * 64 * 4);
  float* t1   = (float*)alloc((size_t)Np * 256 * 4);
  float* t2   = (float*)alloc((size_t)Np * 256 * 4);
  float* allh = (float*)alloc((size_t)Np * 320 * 4);
  float* enc  = (float*)alloc((size_t)Np * 32 * 4);
  float* zf   = (float*)alloc((size_t)Np * 32 * 4);
  float* skey = (float*)alloc((size_t)NS * 4);
  float* psum = (float*)alloc((size_t)G * 32 * 4);
  float* pcnt = (float*)alloc((size_t)G * 4);
  int* sidx = (int*)alloc((size_t)NS * 4);
  int* srt0 = (int*)alloc((size_t)Np * 4);
  int* srt1 = (int*)alloc((size_t)Np * 4);

  auto gemm = [&](const float* A, int lda, const float* B, int ldb,
                  const float* bias, const float* res, float* C, int ldc,
                  int M, int K, int Nn, int act) {
    dim3 grid(Nn / 16, M / 64);
    k_gemm<<<grid, 128, 0, stream>>>(A, lda, B, ldb, bias, res, C, ldc, M, K, Nn, act);
  };

  // ---- pad inputs ----
  k_pad<<<(Np * 12 + 255) / 256, 256, 0, stream>>>(x, coords, xp, c2, N, Np);

  // ---- two coordinate argsorts (bitonic over NS key/index pairs) ----
  for (int d = 0; d < 2; ++d) {
    k_sortinit<<<NS / 256, 256, 0, stream>>>(skey, sidx, coords, N, Np, NS, d);
    for (int k = 2; k <= NS; k <<= 1)
      for (int j = k >> 1; j > 0; j >>= 1)
        k_bitonic<<<NS / 256, 256, 0, stream>>>(skey, sidx, j, k, NS);
    k_take<<<(Np + 255) / 256, 256, 0, stream>>>(d == 0 ? srt0 : srt1, sidx, Np);
  }

  // ---- feature encoder: h = relu(xp@fe1w+fe1b)@fe2w+fe2b ----
  gemm(xp, 12, fe1w, 64, fe1b, nullptr, t1, 64, Np, 12, 64, 1);
  gemm(t1, 64, fe2w, 64, fe2b, nullptr, h, 64, Np, 64, 64, 0);
  k_copy_col<<<(Np * 64 + 255) / 256, 256, 0, stream>>>(h, allh, Np, 0);

  // ---- 4 transformer layers ----
  for (int l = 0; l < 4; ++l) {
    const LayerP& P = L[l];
    k_layernorm<<<Np / 8, 256, 0, stream>>>(h, P.n1g, P.n1b, xn, Np, 64, 0);
    k_attention<<<nb, 256, ATT_SMEM_BYTES, stream>>>(
        xn, c2, srt0, P.wq, P.wk, P.wv, P.wo, P.wrpe, P.wrpeb, h);
    k_attention<<<nb, 256, ATT_SMEM_BYTES, stream>>>(
        xn, c2, srt1, P.wq, P.wk, P.wv, P.wo, P.wrpe, P.wrpeb, h);
    k_layernorm<<<Np / 8, 256, 0, stream>>>(h, P.n2g, P.n2b, xn, Np, 64, 0);
    gemm(xn, 64, P.f1w, 64, P.f1b, nullptr, t1, 64, Np, 64, 64, 1);
    gemm(t1, 64, P.f2w, 64, P.f2b, h, h, 64, Np, 64, 64, 0);
    k_copy_col<<<(Np * 64 + 255) / 256, 256, 0, stream>>>(h, allh, Np, 64 * (l + 1));
  }

  // ---- enc = tanh(allh @ Ww) ----
  gemm(allh, 320, Ww, 32, nullptr, nullptr, enc, 32, Np, 320, 32, 2);

  // ---- MLP: 32 -> 256 x4 -> 32 with tanh(LN) between ----
  gemm(enc, 32, mw[0], 256, mb[0], nullptr, t1, 256, Np, 32, 256, 0);
  k_layernorm<<<Np / 8, 256, 0, stream>>>(t1, mg[0], mnb[0], t1, Np, 256, 1);
  gemm(t1, 256, mw[1], 256, mb[1], nullptr, t2, 256, Np, 256, 256, 0);
  k_layernorm<<<Np / 8, 256, 0, stream>>>(t2, mg[1], mnb[1], t2, Np, 256, 1);
  gemm(t2, 256, mw[2], 256, mb[2], nullptr, t1, 256, Np, 256, 256, 0);
  k_layernorm<<<Np / 8, 256, 0, stream>>>(t1, mg[2], mnb[2], t1, Np, 256, 1);
  gemm(t1, 256, mw[3], 256, mb[3], nullptr, t2, 256, Np, 256, 256, 0);
  k_layernorm<<<Np / 8, 256, 0, stream>>>(t2, mg[3], mnb[3], t2, Np, 256, 1);
  gemm(t2, 256, mw[4], 32, mb[4], nullptr, zf, 32, Np, 256, 32, 0);

  // ---- segment-mean pool + head ----
  k_zero<<<(G * 32 + 255) / 256, 256, 0, stream>>>(psum, G * 32);
  k_zero<<<(G + 255) / 256, 256, 0, stream>>>(pcnt, G);
  k_pool<<<(N * 32 + 255) / 256, 256, 0, stream>>>(enc, zf, batch, psum, pcnt, N);
  k_final<<<(G + 255) / 256, 256, 0, stream>>>(psum, pcnt, opw, opb, (float*)d_out, G);
}
